// FlexSparseGlobalAttention_59219009077403
// MI455X (gfx1250) — matrix operand order repaired and checked by reading, not compiled, not run on hardware
//
#include <hip/hip_runtime.h>
#include <hip/hip_bf16.h>
#include <math.h>

// CDNA5 WMMA fragment types (probe-verified signatures)
typedef __attribute__((ext_vector_type(16))) __bf16 v16bf;
typedef __attribute__((ext_vector_type(8)))  float  v8f;

union Frag {
    v16bf v;
    unsigned short u[16];
    unsigned int d[8];
};

__device__ __forceinline__ unsigned short f2bf(float f) {
    unsigned int u = __float_as_uint(f);
    u += 0x7fffu + ((u >> 16) & 1u);   // round-to-nearest-even
    return (unsigned short)(u >> 16);
}

// ---------------------------------------------------------------------------
// fp32 -> bf16(raw u16) elementwise (activations, row-major kept as-is)
// ---------------------------------------------------------------------------
__global__ void cvt_bf16_kernel(const float* __restrict__ src,
                                unsigned short* __restrict__ dst, int n) {
    int i = blockIdx.x * blockDim.x + threadIdx.x;
    if (i < n) dst[i] = f2bf(src[i]);
}

// ---------------------------------------------------------------------------
// fp32 W[K,N] -> bf16 pair-interleaved uint Bp[K/2][N]:
//   Bp[m*N+n] = bf16(W[2m,n]) | bf16(W[2m+1,n]) << 16
// This makes every WMMA B-fragment VGPR a single aligned 4-byte load.
// ---------------------------------------------------------------------------
__global__ void cvt_pack_kernel(const float* __restrict__ W,
                                unsigned int* __restrict__ Bp,
                                int Khalf, int N) {
    int i = blockIdx.x * blockDim.x + threadIdx.x;
    if (i >= Khalf * N) return;
    int m = i / N, n = i - m * N;
    unsigned int lo = f2bf(W[(size_t)(2 * m) * N + n]);
    unsigned int hi = f2bf(W[(size_t)(2 * m + 1) * N + n]);
    Bp[i] = lo | (hi << 16);
}

// ---------------------------------------------------------------------------
// Repack V section of qkv (bf16 [N, 2304], V at +1536) into pair-interleaved
// uint vp[N/2][768]: vp[m*768+c] = v[2m,c] | v[2m+1,c] << 16.
// kv rows are the WMMA K-dimension of the P@V matmul.
// ---------------------------------------------------------------------------
__global__ void repack_v_kernel(const unsigned short* __restrict__ qkv,
                                unsigned int* __restrict__ vp,
                                int Mhalf, int C) {
    int i = blockIdx.x * blockDim.x + threadIdx.x;
    if (i >= Mhalf * C) return;
    int m = i / C, c = i - m * C;
    unsigned int lo = qkv[(size_t)(2 * m) * 2304 + 1536 + c];
    unsigned int hi = qkv[(size_t)(2 * m + 1) * 2304 + 1536 + c];
    vp[i] = lo | (hi << 16);
}

// ---------------------------------------------------------------------------
// GEMM: out[M,N] = A[M,K](bf16 row-major) @ Bp[K/2,N](bf16 pairs) + bias[N]
// 256 threads = 8 waves; block tile 128(M) x 64(N); wave tile 16 x 64.
// A-layout (16-bit A 16x32): lane<16 -> row=lane, K={0..15} (uint pairs);
//                            lane>=16 -> row=lane-16, K={16..31}.
// B-layout (16-bit B 32x16): col=lane%16; vgpr j -> K pair j (+8 for hi half).
// ---------------------------------------------------------------------------
template <int OUT_BF16>
__global__ __launch_bounds__(256, 1)
void gemm_wmma_kernel(const unsigned short* __restrict__ A,
                      const unsigned int* __restrict__ Bp,
                      const float* __restrict__ bias,
                      void* __restrict__ out,
                      int M, int N, int K) {
    const int wave = threadIdx.x >> 5;
    const int lane = threadIdx.x & 31;
    const int l15  = lane & 15;
    const int hi   = (lane >> 4) & 1;

    const int m0 = blockIdx.y * 128 + wave * 16;
    const int n0 = blockIdx.x * 64;

    v8f c[4];
    for (int t = 0; t < 4; ++t)
        for (int r = 0; r < 8; ++r) c[t][r] = 0.f;

    const size_t arow  = (size_t)(m0 + l15) * K;
    const int    bprow = hi ? 8 : 0;    // pair-row offset for hi half

    for (int k0 = 0; k0 < K; k0 += 32) {
        Frag a;
#pragma unroll
        for (int j = 0; j < 8; ++j) {
            int kk = 2 * j + (hi ? 16 : 0);
            a.d[j] = *(const unsigned int*)(A + arow + k0 + kk);
        }
        const size_t kp = (size_t)(k0 >> 1);
#pragma unroll
        for (int t = 0; t < 4; ++t) {
            Frag b;
            int col = n0 + t * 16 + l15;
#pragma unroll
            for (int j = 0; j < 8; ++j)
                b.d[j] = Bp[(kp + bprow + j) * N + col];
            c[t] = __builtin_amdgcn_wmma_f32_16x16x32_bf16(
                false, a.v, false, b.v, (short)0, c[t], false, false);
        }
    }

#pragma unroll
    for (int t = 0; t < 4; ++t) {
        int col = n0 + t * 16 + l15;
        float bs = bias[col];
#pragma unroll
        for (int r = 0; r < 8; ++r) {
            int row = m0 + r + (hi ? 8 : 0);
            float v = c[t][r] + bs;
            if (OUT_BF16)
                ((unsigned short*)out)[(size_t)row * N + col] = f2bf(v);
            else
                ((float*)out)[(size_t)row * N + col] = v;
        }
    }
}

// ---------------------------------------------------------------------------
// Block-sparse flash attention.
// Frame mask (S=24, window=4, anchor_step=3) is 128x128-block granular:
// kv frame kf allowed for query frame qf iff |qf-kf|<=4 or kf%3==0.
// One block per (head, q-frame); 8 waves; wave owns 16 q rows x Dh=64.
// Each allowed kv frame is processed as two 64-column flash blocks to keep
// live score tiles at 4 (32 VGPRs). __launch_bounds__(256,1) gives the
// compiler the full VGPR budget so nothing spills to scratch.
// ---------------------------------------------------------------------------
__global__ __launch_bounds__(256, 1)
void attn_wmma_kernel(const unsigned short* __restrict__ qkv,
                      const unsigned int* __restrict__ vp,
                      unsigned short* __restrict__ attn_out) {
    __shared__ unsigned short lds_p[8][16][64];   // 16 KB, wave-private slabs

    const int h    = blockIdx.x;   // 0..11
    const int qf   = blockIdx.y;   // 0..23
    const int wave = threadIdx.x >> 5;
    const int lane = threadIdx.x & 31;
    const int l15  = lane & 15;
    const int hi   = (lane >> 4) & 1;

    const int    LDQ  = 2304;
    const size_t qoff = (size_t)h * 64;
    const size_t koff = 768 + (size_t)h * 64;
    const size_t voff = (size_t)h * 64;     // within vp row (width 768)

    const int   qrow0 = qf * 128 + wave * 16;
    const float scale = 0.125f;             // Dh=64 -> 1/8

    // Q A-fragments, K = Dh split into 0..31 / 32..63
    Frag aq[2];
#pragma unroll
    for (int f = 0; f < 2; ++f) {
#pragma unroll
        for (int j = 0; j < 8; ++j) {
            int kk = 2 * j + (hi ? 16 : 0);
            aq[f].d[j] = *(const unsigned int*)(
                qkv + (size_t)(qrow0 + l15) * LDQ + qoff + f * 32 + kk);
        }
    }

    float mrow[8], lrow[8];
    v8f acc[4];
    for (int r = 0; r < 8; ++r) { mrow[r] = -1e30f; lrow[r] = 0.f; }
    for (int t = 0; t < 4; ++t)
        for (int r = 0; r < 8; ++r) acc[t][r] = 0.f;

    for (int kf = 0; kf < 24; ++kf) {
        int  dd      = qf - kf;
        bool allowed = (dd >= -4 && dd <= 4) || ((kf % 3) == 0);
        if (!allowed) continue;

        for (int half = 0; half < 2; ++half) {
            const int cb = kf * 128 + half * 64;   // kv col base of this block

            // S block: q(16x64) @ k^T(64x64) -> 4 column tiles of 16
            v8f s[4];
#pragma unroll
            for (int ct = 0; ct < 4; ++ct) {
                const size_t krow = (size_t)(cb + ct * 16 + l15) * LDQ + koff;
                Frag b0, b1;
#pragma unroll
                for (int j = 0; j < 8; ++j) {
                    int kk = 2 * j + (hi ? 16 : 0);
                    b0.d[j] = *(const unsigned int*)(qkv + krow + kk);
                    b1.d[j] = *(const unsigned int*)(qkv + krow + 32 + kk);
                }
                v8f z;
                for (int r = 0; r < 8; ++r) z[r] = 0.f;
                z = __builtin_amdgcn_wmma_f32_16x16x32_bf16(
                    false, aq[0].v, false, b0.v, (short)0, z, false, false);
                z = __builtin_amdgcn_wmma_f32_16x16x32_bf16(
                    false, aq[1].v, false, b1.v, (short)0, z, false, false);
#pragma unroll
                for (int r = 0; r < 8; ++r) s[ct][r] = z[r] * scale;
            }

            // online softmax: row max across tiles + across each 16-lane half
            float corr[8];
#pragma unroll
            for (int r = 0; r < 8; ++r) {
                float m = s[0][r];
#pragma unroll
                for (int ct = 1; ct < 4; ++ct) m = fmaxf(m, s[ct][r]);
                m = fmaxf(m, __shfl_xor(m, 1, 32));
                m = fmaxf(m, __shfl_xor(m, 2, 32));
                m = fmaxf(m, __shfl_xor(m, 4, 32));
                m = fmaxf(m, __shfl_xor(m, 8, 32));
                float mn = fmaxf(mrow[r], m);
                corr[r]  = __expf(mrow[r] - mn);
                mrow[r]  = mn;
            }

            // P = exp(S - m) -> LDS (bf16, C-layout addressing); row sums
            float ts[8];
            for (int r = 0; r < 8; ++r) ts[r] = 0.f;
#pragma unroll
            for (int ct = 0; ct < 4; ++ct) {
#pragma unroll
                for (int r = 0; r < 8; ++r) {
                    float p = __expf(s[ct][r] - mrow[r]);
                    ts[r] += p;
                    lds_p[wave][r + (hi ? 8 : 0)][ct * 16 + l15] = f2bf(p);
                }
            }
#pragma unroll
            for (int r = 0; r < 8; ++r) {
                float t = ts[r];
                t += __shfl_xor(t, 1, 32);
                t += __shfl_xor(t, 2, 32);
                t += __shfl_xor(t, 4, 32);
                t += __shfl_xor(t, 8, 32);
                lrow[r] = lrow[r] * corr[r] + t;
            }
#pragma unroll
            for (int t = 0; t < 4; ++t)
                for (int r = 0; r < 8; ++r) acc[t][r] *= corr[r];

            // RAW on wave-private LDS slab: drain DS counter before re-read
            asm volatile("s_wait_dscnt 0" ::: "memory");

            // P A-fragments (K = 64 kv positions -> 2 fragments of 32)
            Frag pa[2];
#pragma unroll
            for (int f = 0; f < 2; ++f) {
#pragma unroll
                for (int j = 0; j < 8; ++j) {
                    int kk = 2 * j + (hi ? 16 : 0);
                    pa[f].d[j] =
                        *(const unsigned int*)&lds_p[wave][l15][f * 32 + kk];
                }
            }

            // acc += P(16x64) @ V(64x64); V pair-interleaved -> 1 uint/VGPR
            const size_t vprow0 = (size_t)(cb >> 1) + (hi ? 8 : 0);
#pragma unroll
            for (int dt = 0; dt < 4; ++dt) {
                int col = dt * 16 + l15;
#pragma unroll
                for (int f = 0; f < 2; ++f) {
                    Frag bv;
#pragma unroll
                    for (int j = 0; j < 8; ++j)
                        bv.d[j] = vp[(vprow0 + f * 16 + j) * 768 + voff + col];
                    acc[dt] = __builtin_amdgcn_wmma_f32_16x16x32_bf16(
                        false, pa[f].v, false, bv.v, (short)0, acc[dt],
                        false, false);
                }
            }
        }
    }

    // normalize and emit attention output (bf16) in [N, C] at column h*64+d
#pragma unroll
    for (int dt = 0; dt < 4; ++dt) {
        int d = dt * 16 + l15;
#pragma unroll
        for (int r = 0; r < 8; ++r) {
            int row = qrow0 + r + (hi ? 8 : 0);
            attn_out[(size_t)row * 768 + h * 64 + d] = f2bf(acc[dt][r] / lrow[r]);
        }
    }
}

// ---------------------------------------------------------------------------
extern "C" void kernel_launch(void* const* d_in, const int* in_sizes, int n_in,
                              void* d_out, int out_size, void* d_ws, size_t ws_size,
                              hipStream_t stream) {
    (void)in_sizes; (void)n_in; (void)out_size; (void)ws_size;

    const float* x      = (const float*)d_in[0];  // [1,3072,768]
    const float* W_qkv  = (const float*)d_in[1];  // [768,2304]
    const float* b_qkv  = (const float*)d_in[2];  // [2304]
    const float* W_proj = (const float*)d_in[3];  // [768,768]
    const float* b_proj = (const float*)d_in[4];  // [768]

    const int N = 3072, C = 768, C3 = 2304;

    char*  ws  = (char*)d_ws;
    size_t off = 0;
    auto alloc = [&](size_t bytes) -> void* {
        void* p = ws + off;
        off = (off + bytes + 255) & ~(size_t)255;
        return p;
    };
    unsigned short* xb     = (unsigned short*)alloc((size_t)N * C * 2);
    unsigned int*   wqkvp  = (unsigned int*)alloc((size_t)(C / 2) * C3 * 4);
    unsigned int*   wprojp = (unsigned int*)alloc((size_t)(C / 2) * C * 4);
    unsigned short* qkvb   = (unsigned short*)alloc((size_t)N * C3 * 2);
    unsigned int*   vpack  = (unsigned int*)alloc((size_t)(N / 2) * C * 4);
    unsigned short* attnb  = (unsigned short*)alloc((size_t)N * C * 2);

    // 1) fp32 -> bf16 conversions (x row-major; weights pair-interleaved)
    {
        int n = N * C;
        cvt_bf16_kernel<<<(n + 255) / 256, 256, 0, stream>>>(x, xb, n);
        n = (C / 2) * C3;
        cvt_pack_kernel<<<(n + 255) / 256, 256, 0, stream>>>(W_qkv, wqkvp,
                                                             C / 2, C3);
        n = (C / 2) * C;
        cvt_pack_kernel<<<(n + 255) / 256, 256, 0, stream>>>(W_proj, wprojp,
                                                             C / 2, C);
    }

    // 2) QKV projection: [3072,768] @ [768,2304] + bias -> bf16 qkv
    {
        dim3 grid(C3 / 64, N / 128);
        gemm_wmma_kernel<1><<<grid, 256, 0, stream>>>(xb, wqkvp, b_qkv, qkvb,
                                                      N, C3, C);
    }

    // 3) pair-interleave V rows for the P@V B-fragments
    {
        int n = (N / 2) * C;
        repack_v_kernel<<<(n + 255) / 256, 256, 0, stream>>>(qkvb, vpack,
                                                             N / 2, C);
    }

    // 4) block-sparse flash attention: one block per (head, q-frame)
    {
        dim3 grid(12, 24);
        attn_wmma_kernel<<<grid, 256, 0, stream>>>(qkvb, vpack, attnb);
    }

    // 5) output projection: [3072,768] @ [768,768] + bias -> fp32 d_out
    {
        dim3 grid(C / 64, N / 128);
        gemm_wmma_kernel<0><<<grid, 256, 0, stream>>>(attnb, wprojp, b_proj,
                                                      d_out, N, C, C);
    }
}
